// GraphMixerV7_43662637532067
// MI455X (gfx1250) — matrix operand order repaired and checked by reading, not compiled, not run on hardware
//
#include <hip/hip_runtime.h>
#include <hip/hip_bf16.h>

// ---------------------------------------------------------------------------
// GraphMixer on MI455X (gfx1250).
// GEMMs: v_wmma_f32_16x16x32_bf16, one block = 8 waves sharing a 16-row A tile
// staged into LDS with GLOBAL_LOAD_ASYNC_TO_LDS_B128 (ASYNCcnt path).
// Shapes: B=8, N=128, T=64, D=512, H=8, hd=64, TOPK=16, SCALE=0.125.
// Workspace (bytes), total ~137 MB:
//   wq/wk/wv/wo bf16 : 4 x 512KB   @ 0, 512K, 1M, 1.5M
//   z_bf  [1024,512] : 1 MB        @ 2M
//   q32   [1024,512] : 2 MB        @ 3M
//   k32   [1024,512] : 2 MB        @ 5M
//   attn  [B,H,N,N]  : 2 MB bf16   @ 7M
//   vt    [B,H,4096,128] bf16 64MB @ 9M      (V stored (t,d)-major, j-contig)
//   mixed [B,N,T,D]  bf16 64MB     @ 73M
// ---------------------------------------------------------------------------

typedef __attribute__((ext_vector_type(16))) __bf16 v16bf;
typedef __attribute__((ext_vector_type(8)))  float  v8f;

union Frag {
  v16bf v;
  unsigned short u[16];
  uint4 q[2];
};

__device__ __forceinline__ unsigned short f2bf(float f) {
  unsigned u = __float_as_uint(f);
  unsigned r = u + 0x7FFFu + ((u >> 16) & 1u);   // RNE
  return (unsigned short)(r >> 16);
}

// Async copy 16 bytes global -> LDS (ASYNCcnt-tracked, no VGPR data).
__device__ __forceinline__ void async_b128(unsigned lds_off, const void* g) {
  unsigned long long ga = (unsigned long long)g;
  asm volatile("global_load_async_to_lds_b128 %0, %1, off"
               :: "v"(lds_off), "v"(ga) : "memory");
}
__device__ __forceinline__ void async_wait0() {
  asm volatile("s_wait_asynccnt 0" ::: "memory");
}

// One 16x16 tile of an NT GEMM: D = A[16xK] * B[16xK]^T, bf16 in, f32 acc.
// A lives in LDS (padded leading dim LDA), B streamed from global (ld LDB).
// A frag (16x32): lane M=lane&15, half=lane>>4; elems 0..7 -> K=half*8+i,
// elems 8..15 -> K=16+half*8+i.  B frag (32x16): lane N=lane&15, same K order.
template<int K, int LDA, int LDB>
__device__ __forceinline__ v8f wave_tile(const unsigned short* __restrict__ aLds,
                                         const unsigned short* __restrict__ B,
                                         int lane) {
  v8f acc = {};
  const int M  = lane & 15;
  const int hi = lane >> 4;
  const unsigned short* ap = aLds + M * LDA + hi * 8;
  const unsigned short* bp = B + M * LDB + hi * 8;
#pragma unroll
  for (int k0 = 0; k0 < K; k0 += 32) {
    if (k0 + 64 <= K) __builtin_prefetch(bp + k0 + 32, 0, 0);  // global_prefetch_b8
    Frag fa, fb;
    fa.q[0] = *(const uint4*)(ap + k0);          // ds_load_b128
    fa.q[1] = *(const uint4*)(ap + k0 + 16);
    fb.q[0] = *(const uint4*)(bp + k0);          // global_load_b128
    fb.q[1] = *(const uint4*)(bp + k0 + 16);
    acc = __builtin_amdgcn_wmma_f32_16x16x32_bf16(false, fa.v, false, fb.v,
                                                  (short)0, acc, false, false);
  }
  return acc;
}

// ---------------------------------------------------------------------------
__global__ void k_cvt_bf16(const float* __restrict__ src,
                           unsigned short* __restrict__ dst, int n) {
  int i = blockIdx.x * blockDim.x + threadIdx.x;
  if (i < n) dst[i] = f2bf(src[i]);
}

__global__ void k_zero(float* __restrict__ p, int n) {
  int i = blockIdx.x * blockDim.x + threadIdx.x;
  if (i < n) p[i] = 0.f;
}

// z_map = mean over T of h_map : [B,N,T,D] -> bf16 [B*N, D]
__global__ void k_zmap(const float* __restrict__ h_map,
                       unsigned short* __restrict__ z_bf) {
  int idx = blockIdx.x * blockDim.x + threadIdx.x;   // 524288
  int d  = idx & 511;
  int bn = idx >> 9;
  const float* p = h_map + ((size_t)bn * 64) * 512 + d;
  float s = 0.f;
#pragma unroll 8
  for (int t = 0; t < 64; ++t) s += p[(size_t)t * 512];
  z_bf[idx] = f2bf(s * (1.f / 64.f));
}

// q = z @ Wq^T, k = z @ Wk^T : [1024,512]x[512,512] NT.
// Block: 8 waves share one 16x512 A tile (async-staged), cover 8 n-tiles.
__global__ __launch_bounds__(256)
void k_qkproj(const unsigned short* __restrict__ z_bf,
              const unsigned short* __restrict__ wq,
              const unsigned short* __restrict__ wk,
              float* __restrict__ q32, float* __restrict__ k32) {
  extern __shared__ char smem[];                  // 16 x 520 x 2 = 16640 B
  int bx = blockIdx.x;                            // 512 blocks
  int which = bx >> 8;                            // 0 -> q, 1 -> k
  int rem   = bx & 255;
  int mt    = rem >> 2;                           // 0..63
  int nblk  = rem & 3;
  const char* gsrc = (const char*)(z_bf + (size_t)mt * 16 * 512);
#pragma unroll
  for (int j = 0; j < 4; ++j) {
    unsigned off = threadIdx.x * 16 + j * 4096;   // 16KB tile
    unsigned lds = off + ((off >> 10) << 4);      // +16B pad per 1024B row
    async_b128(lds, gsrc + off);
  }
  async_wait0();
  __syncthreads();
  int w    = threadIdx.x >> 5;
  int lane = threadIdx.x & 31;
  int nt   = nblk * 8 + w;
  const unsigned short* W = which ? wk : wq;
  v8f acc = wave_tile<512, 520, 512>((const unsigned short*)smem,
                                     W + (size_t)nt * 16 * 512, lane);
  float* out = which ? k32 : q32;
  int m0  = mt * 16 + ((lane >> 4) << 3);
  int col = nt * 16 + (lane & 15);
#pragma unroll
  for (int r = 0; r < 8; ++r) out[(size_t)(m0 + r) * 512 + col] = acc[r];
}

// scores + log-bias, raw softmax mean, top-16 masked softmax -> attn_bf + means.
__global__ __launch_bounds__(128)
void k_scores(const float* __restrict__ q32, const float* __restrict__ k32,
              const float* __restrict__ base_adj,
              unsigned short* __restrict__ attn_bf,
              float* __restrict__ attn_mean, float* __restrict__ raw_mean) {
  int blk = blockIdx.x;                 // b*H*N + h*N + i  (8192)
  int i  = blk & 127;
  int bh = blk >> 7;
  int h  = bh & 7;
  int b  = bh >> 3;
  int j  = threadIdx.x;                 // 0..127

  __shared__ float qrow[64];
  __shared__ float sc[128];
  __shared__ float ex[128];
  __shared__ int   keep[128];

  if (j < 64) qrow[j] = q32[((size_t)(b * 128 + i)) * 512 + h * 64 + j];
  __syncthreads();

  const float* krow = k32 + ((size_t)(b * 128 + j)) * 512 + h * 64;
  float s = 0.f;
#pragma unroll 8
  for (int d = 0; d < 64; ++d) s += qrow[d] * krow[d];
  float ba = fmaxf(base_adj[((size_t)b * 128 + i) * 128 + j], 1e-12f);
  s = s * 0.125f + __logf(ba);          // SCALE = 1/sqrt(64), BIAS_BASE = 1
  sc[j] = s;
  __syncthreads();

  float mx = -3.4e38f;
  for (int l = 0; l < 128; ++l) mx = fmaxf(mx, sc[l]);
  float e = __expf(s - mx);
  int rank = 0;
  for (int l = 0; l < 128; ++l) {
    float sl = sc[l];
    rank += (sl > s) || (sl == s && l < j);   // top_k first-occurrence ties
  }
  int kp = rank < 16;
  ex[j] = e; keep[j] = kp;
  __syncthreads();

  float rsum = 0.f, msum = 0.f;
  for (int l = 0; l < 128; ++l) { rsum += ex[l]; msum += keep[l] ? ex[l] : 0.f; }
  float raw = e / rsum;
  float at  = kp ? (e / msum) : 0.f;

  attn_bf[(((size_t)bh) * 128 + i) * 128 + j] = f2bf(at);
  atomicAdd(&attn_mean[((size_t)b * 128 + i) * 128 + j], at  * 0.125f);
  atomicAdd(&raw_mean [((size_t)b * 128 + i) * 128 + j], raw * 0.125f);
}

// v = h_val @ Wv^T, output transposed per (b,h): vt[(b,h)][t*64+dh][j=nseg].
// fp32 A tile async-staged raw into LDS, converted once to bf16, shared by 8 waves.
__global__ __launch_bounds__(256)
void k_vproj(const float* __restrict__ h_val,
             const unsigned short* __restrict__ wvw,
             unsigned short* __restrict__ vt) {
  extern __shared__ char smem[];                  // 32768 fp32 + 16640 bf16
  int bx   = blockIdx.x;                          // 16384 blocks
  int b    = bx >> 11;
  int rem  = bx & 2047;
  int mt   = rem >> 2;                            // 0..511
  int nblk = rem & 3;
  const char* gsrc = (const char*)(h_val + ((size_t)b * 8192 + mt * 16) * 512);
#pragma unroll
  for (int j = 0; j < 8; ++j) {
    unsigned off = threadIdx.x * 16 + j * 4096;   // 32KB fp32 tile
    async_b128(off, gsrc + off);
  }
  async_wait0();
  __syncthreads();
  const float* af = (const float*)smem;
  unsigned short* ab = (unsigned short*)(smem + 32768);
#pragma unroll
  for (int j = 0; j < 32; ++j) {
    int i = threadIdx.x + j * 256;
    ab[(i >> 9) * 520 + (i & 511)] = f2bf(af[i]);
  }
  __syncthreads();
  int w    = threadIdx.x >> 5;
  int lane = threadIdx.x & 31;
  int nt   = nblk * 8 + w;
  v8f acc = wave_tile<512, 520, 512>(ab, wvw + (size_t)nt * 16 * 512, lane);
  int col = nt * 16 + (lane & 15);                // h*64 + dh
  int h   = col >> 6, dh = col & 63;
  int m0  = mt * 16 + ((lane >> 4) << 3);
#pragma unroll
  for (int r = 0; r < 8; ++r) {
    int m = m0 + r;
    int nseg = m >> 6, t = m & 63;
    vt[(((size_t)(b * 8 + h)) * 4096 + (t * 64 + dh)) * 128 + nseg] = f2bf(acc[r]);
  }
}

// mixed[b,h,i,(t,dh)] = attn[b,h] @ vt[b,h] : [128,128]x[128,4096] NT-style.
__global__ __launch_bounds__(256)
void k_mix(const unsigned short* __restrict__ attn_bf,
           const unsigned short* __restrict__ vt,
           unsigned short* __restrict__ mixed) {
  extern __shared__ char smem[];                  // 16 x 136 x 2 = 4352 B
  int bx   = blockIdx.x;                          // 16384 blocks
  int bh   = bx >> 8;
  int rem  = bx & 255;
  int mt   = rem >> 5;                            // 0..7
  int nblk = rem & 31;                            // 0..31
  const char* gsrc = (const char*)(attn_bf + (((size_t)bh) * 128 + mt * 16) * 128);
  {
    unsigned off = threadIdx.x * 16;              // 4KB tile, 1 chunk/thread
    unsigned lds = off + ((off >> 8) << 4);       // +16B pad per 256B row
    async_b128(lds, gsrc + off);
  }
  async_wait0();
  __syncthreads();
  int w    = threadIdx.x >> 5;
  int lane = threadIdx.x & 31;
  int nt   = nblk * 8 + w;                        // 0..255
  v8f acc = wave_tile<128, 136, 128>((const unsigned short*)smem,
                                     vt + (((size_t)bh) * 4096 + nt * 16) * 128, lane);
  int b = bh >> 3, h = bh & 7;
  int col2 = nt * 16 + (lane & 15);               // t*64 + dh
  int t = col2 >> 6, dh = col2 & 63;
  int i0 = mt * 16 + ((lane >> 4) << 3);
#pragma unroll
  for (int r = 0; r < 8; ++r) {
    int i = i0 + r;
    mixed[(((size_t)(b * 128 + i)) * 64 + t) * 512 + h * 64 + dh] = f2bf(acc[r]);
  }
}

// out = h_val + clamp(residual_scale[n]) * (mixed @ Wo^T)
__global__ __launch_bounds__(256)
void k_oproj(const unsigned short* __restrict__ mixed,
             const unsigned short* __restrict__ wo,
             const float* __restrict__ h_val,
             const float* __restrict__ rs,
             float* __restrict__ out) {
  extern __shared__ char smem[];                  // 16 x 520 x 2 = 16640 B
  int bx   = blockIdx.x;                          // 16384 blocks
  int mt   = bx >> 2;                             // 0..4095
  int nblk = bx & 3;
  const char* gsrc = (const char*)(mixed + (size_t)mt * 16 * 512);
#pragma unroll
  for (int j = 0; j < 4; ++j) {
    unsigned off = threadIdx.x * 16 + j * 4096;
    unsigned lds = off + ((off >> 10) << 4);
    async_b128(lds, gsrc + off);
  }
  async_wait0();
  __syncthreads();
  int w    = threadIdx.x >> 5;
  int lane = threadIdx.x & 31;
  int nt   = nblk * 8 + w;
  v8f acc = wave_tile<512, 520, 512>((const unsigned short*)smem,
                                     wo + (size_t)nt * 16 * 512, lane);
  int col = nt * 16 + (lane & 15);
  int m0  = mt * 16 + ((lane >> 4) << 3);
#pragma unroll
  for (int r = 0; r < 8; ++r) {
    int row = m0 + r;
    int n = (row >> 6) & 127;
    float s = fminf(fmaxf(rs[n], 0.f), 1.f);
    size_t idx = (size_t)row * 512 + col;
    out[idx] = h_val[idx] + acc[r] * s;
  }
}

// ---------------------------------------------------------------------------
extern "C" void kernel_launch(void* const* d_in, const int* in_sizes, int n_in,
                              void* d_out, int out_size, void* d_ws, size_t ws_size,
                              hipStream_t stream) {
  (void)in_sizes; (void)n_in; (void)out_size; (void)ws_size;

  const float* h_val    = (const float*)d_in[0];
  const float* h_map    = (const float*)d_in[1];
  const float* base_adj = (const float*)d_in[2];
  const float* q_w      = (const float*)d_in[3];
  const float* k_w      = (const float*)d_in[4];
  const float* v_w      = (const float*)d_in[5];
  const float* o_w      = (const float*)d_in[6];
  const float* rs       = (const float*)d_in[7];

  float* out       = (float*)d_out;              // [B,N,T,D] = 33554432
  float* attn_mean = out + 33554432;             // [B,N,N]   = 131072
  float* raw_mean  = attn_mean + 131072;         // [B,N,N]   = 131072

  char* ws = (char*)d_ws;
  unsigned short* wq_bf  = (unsigned short*)(ws + 0);
  unsigned short* wk_bf  = (unsigned short*)(ws + 524288);
  unsigned short* wv_bf  = (unsigned short*)(ws + 1048576);
  unsigned short* wo_bf  = (unsigned short*)(ws + 1572864);
  unsigned short* z_bf   = (unsigned short*)(ws + 2097152);
  float*          q32    = (float*)         (ws + 3145728);
  float*          k32    = (float*)         (ws + 5242880);
  unsigned short* attn   = (unsigned short*)(ws + 7340032);
  unsigned short* vt     = (unsigned short*)(ws + 9437184);
  unsigned short* mixed  = (unsigned short*)(ws + 76546048);

  // 1. weights -> bf16
  k_cvt_bf16<<<1024, 256, 0, stream>>>(q_w, wq_bf, 262144);
  k_cvt_bf16<<<1024, 256, 0, stream>>>(k_w, wk_bf, 262144);
  k_cvt_bf16<<<1024, 256, 0, stream>>>(v_w, wv_bf, 262144);
  k_cvt_bf16<<<1024, 256, 0, stream>>>(o_w, wo_bf, 262144);
  // 2. z_map mean + cvt
  k_zmap<<<2048, 256, 0, stream>>>(h_map, z_bf);
  // 3. zero the attention-mean accumulators in d_out tail
  k_zero<<<1024, 256, 0, stream>>>(attn_mean, 262144);
  // 4. q/k projections (WMMA, async-LDS A tiles)
  k_qkproj<<<512, 256, 16640, stream>>>(z_bf, wq_bf, wk_bf, q32, k32);
  // 5. scores + softmax + top-16 + head means
  k_scores<<<8192, 128, 0, stream>>>(q32, k32, base_adj, attn, attn_mean, raw_mean);
  // 6. v projection (WMMA, fp32 tile async-staged + converted once per block)
  k_vproj<<<16384, 256, 49408, stream>>>(h_val, wv_bf, vt);
  // 7. attn @ V mixing (WMMA)
  k_mix<<<16384, 256, 4352, stream>>>(attn, vt, mixed);
  // 8. o projection + clamped residual (WMMA)
  k_oproj<<<16384, 256, 16640, stream>>>(mixed, wo_bf, h_val, rs, out);
}
